// TDKNN_net_12953621364879
// MI455X (gfx1250) — matrix operand order repaired and checked by reading, not compiled, not run on hardware
//
#include <hip/hip_runtime.h>
#include <hip/hip_bf16.h>

typedef __bf16 v16bf __attribute__((ext_vector_type(16)));
typedef float  v8f   __attribute__((ext_vector_type(8)));

#define S_DIM 384
#define R_DIM 64
#define K_DIM 16

// ---------------------------------------------------------------------------
// Fragment-packed layouts: each 16x16-output WMMA operand tile is stored as
// [tile][lane 0..31][16 bf16] = 1KB, so a lane's fragment is 32 contiguous
// bytes (two b128 loads) and a wave's tile read is 1KB fully coalesced.
// ---------------------------------------------------------------------------

// A-operand (M x K, 16x32 tiles): lane = kh*16 + m; e -> K = kh*16 + e.
__device__ __forceinline__ int a_pack_off(int m, int k, int nkt) {
  const int mt = m >> 4, mi = m & 15;
  const int kt = k >> 5, kk = k & 31;
  const int kh = (kk >> 4) & 1;
  const int e  = kk & 15;
  return (((mt * nkt + kt) * 32) + (kh * 16 + mi)) * 16 + e;
}

// B-operand (K x N, 32x16 tiles): lane = kh*16 + n; e -> K = kh*16 + e.
__device__ __forceinline__ int b_pack_off(int k, int n, int nkt) {
  const int kt = k >> 5;
  const int kh = (k >> 4) & 1;
  const int e  = k & 15;
  const int nt = n >> 4;
  return (((nt * nkt + kt) * 32) + (kh * 16 + (n & 15))) * 16 + e;
}

__device__ __forceinline__ v16bf load_frag(const __bf16* base, int tile,
                                           int lane) {
  return *(const v16bf*)(base + (tile * 32 + lane) * 16);
}

__device__ __forceinline__ v8f wmma_bf16(v16bf a, v16bf b, v8f c) {
  return __builtin_amdgcn_wmma_f32_16x16x32_bf16(false, a, false, b,
                                                 (short)0, c, false, false);
}

// C/D: VGPR r -> (M = r | 8+r for lane<16 | lane>=16, N = lane&15)
__device__ __forceinline__ int cd_row(int r, int lane) {
  return (lane < 16) ? r : (8 + r);
}

// ---------------------------------------------------------------------------
// K1: G (64^3 fp32, a-major) -> bf16 B-packed (K=a 64, N=b*64+c 4096)
// ---------------------------------------------------------------------------
__global__ void tdknn_convert_g(const float* __restrict__ G,
                                __bf16* __restrict__ Gbp, int n) {
  for (int i = blockIdx.x * blockDim.x + threadIdx.x; i < n;
       i += gridDim.x * blockDim.x) {
    const int a = i >> 12;     // K index
    const int nn = i & 4095;   // N index
    Gbp[b_pack_off(a, nn, 2)] = (__bf16)G[i];
  }
}

// ---------------------------------------------------------------------------
// K2: gather-weighted sums, stored fragment-packed:
//   Mx0, Mx1 -> A-packed (384x64);  Mx2 -> B-packed transposed (K=c, N=k row)
// ---------------------------------------------------------------------------
__global__ void tdknn_gather(const float* __restrict__ M0,
                             const float* __restrict__ M1,
                             const float* __restrict__ M2,
                             const int* __restrict__ idx0,
                             const int* __restrict__ idx1,
                             const int* __restrict__ idx2,
                             const float* __restrict__ w0,
                             const float* __restrict__ w1,
                             const float* __restrict__ w2,
                             __bf16* __restrict__ Mx0p,
                             __bf16* __restrict__ Mx1p,
                             __bf16* __restrict__ Mx2Tp) {
  const int s     = blockIdx.x;   // row 0..383
  const int which = blockIdx.y;   // matrix 0..2
  const int a     = threadIdx.x;  // column 0..63

  const float* M   = (which == 0) ? M0 : (which == 1) ? M1 : M2;
  const int*   idx = (which == 0) ? idx0 : (which == 1) ? idx1 : idx2;
  const float* w   = (which == 0) ? w0 : (which == 1) ? w1 : w2;

  float acc = 0.0f;
#pragma unroll
  for (int k = 0; k < K_DIM; ++k) {
    const int j = idx[s * K_DIM + k];
    acc += M[j * R_DIM + a] * w[s * K_DIM + k];
  }
  if (which == 0)      Mx0p[a_pack_off(s, a, 2)]  = (__bf16)acc;
  else if (which == 1) Mx1p[a_pack_off(s, a, 2)]  = (__bf16)acc;
  else                 Mx2Tp[b_pack_off(a, s, 2)] = (__bf16)acc;
}

// ---------------------------------------------------------------------------
// K3: T1 = Mx0 (384x64) @ G (64x4096); result stored per-i B-packed
//     (for row i: K = b (64), N = c (64), nkt=2 -> 4096 bf16 per i).
// grid = (16 nblocks of 256 cols, 3 mblocks of 128 rows); 8 waves/block.
// ---------------------------------------------------------------------------
__global__ void tdknn_gemm1(const __bf16* __restrict__ Mx0p,
                            const __bf16* __restrict__ Gbp,
                            __bf16* __restrict__ T1p) {
  __shared__ __align__(32) __bf16 smA[8192];  // 16 KB, 16 packed tiles

  const int lane = threadIdx.x & 31;
  const int wave = threadIdx.x >> 5;
  const int nb   = blockIdx.x;  // 0..15
  const int mb   = blockIdx.y;  // 0..2

  {  // contiguous copy of this block's 16 A-tiles
    const uint4* src = (const uint4*)(Mx0p + mb * 8192);
    uint4* dst = (uint4*)smA;
    for (int t = threadIdx.x; t < 1024; t += 256) dst[t] = src[t];
  }
  __syncthreads();

  const v16bf a0 = load_frag(smA, wave * 2 + 0, lane);
  const v16bf a1 = load_frag(smA, wave * 2 + 1, lane);
  const int row0 = mb * 128 + wave * 16;

#pragma unroll 4
  for (int nt = 0; nt < 16; ++nt) {
    const int gnt = nb * 16 + nt;                 // global n-tile 0..255
    const v16bf b0 = load_frag(Gbp, gnt * 2 + 0, lane);
    const v16bf b1 = load_frag(Gbp, gnt * 2 + 1, lane);
    v8f acc = {};
    acc = wmma_bf16(a0, b0, acc);
    acc = wmma_bf16(a1, b1, acc);

    // scatter into per-i B-packed T1: element (i, b, c), n0 = gnt*16
    const int n0 = gnt * 16;
    const int b  = n0 >> 6;           // constant over the 16-wide tile
    const int ct = (n0 >> 4) & 3;     // c-tile
    const int kt = b >> 5, kh = (b >> 4) & 1, e = b & 15;
    const int baseoff =
        (((ct * 2 + kt) * 32) + (kh * 16 + (lane & 15))) * 16 + e;
#pragma unroll
    for (int r = 0; r < 8; ++r)
      T1p[(size_t)(row0 + cd_row(r, lane)) * 4096 + baseoff] = (__bf16)acc[r];
  }
}

// ---------------------------------------------------------------------------
// K4 (fused): block (i, jb):
//   Step A: T2tile (128 j x 64 c) = Mx1[j-block] @ T1_i  -> LDS, A-packed
//   Step B: out[i, j-block, :] = T2tile @ Mx2T           -> fp32 HBM
// ---------------------------------------------------------------------------
__global__ void tdknn_fused_tail(const __bf16* __restrict__ Mx1p,
                                 const __bf16* __restrict__ T1p,
                                 const __bf16* __restrict__ Mx2Tp,
                                 float* __restrict__ out) {
  __shared__ __align__(32) __bf16 smMx1[8192];  // 16 packed A tiles
  __shared__ __align__(32) __bf16 smT1[4096];   //  8 packed B tiles
  __shared__ __align__(32) __bf16 smT2[8192];   // 16 packed A tiles

  const int lane = threadIdx.x & 31;
  const int wave = threadIdx.x >> 5;
  const int jb   = blockIdx.x;  // 0..2
  const int i    = blockIdx.y;  // 0..383
  const int j0   = jb * 128;

  __builtin_prefetch(Mx2Tp + (threadIdx.x << 6), 0, 1);

  {  // contiguous wide copies into LDS
    const uint4* s0 = (const uint4*)(Mx1p + jb * 8192);
    uint4* d0 = (uint4*)smMx1;
    for (int t = threadIdx.x; t < 1024; t += 256) d0[t] = s0[t];
    const uint4* s1 = (const uint4*)(T1p + (size_t)i * 4096);
    uint4* d1 = (uint4*)smT1;
    for (int t = threadIdx.x; t < 512; t += 256) d1[t] = s1[t];
  }
  __syncthreads();

  // ---- Step A: 128x64 T2 tile over K=b, write A-packed into smT2 ----
  {
    const v16bf a0 = load_frag(smMx1, wave * 2 + 0, lane);
    const v16bf a1 = load_frag(smMx1, wave * 2 + 1, lane);
#pragma unroll
    for (int ct = 0; ct < 4; ++ct) {
      const v16bf b0 = load_frag(smT1, ct * 2 + 0, lane);
      const v16bf b1 = load_frag(smT1, ct * 2 + 1, lane);
      v8f acc = {};
      acc = wmma_bf16(a0, b0, acc);
      acc = wmma_bf16(a1, b1, acc);

      // A-pack scatter: m = wave*16 + cd_row, K = c = ct*16 + (lane&15)
      const int n_in = lane & 15;
      const int kt = ct >> 1;
      const int kh = ct & 1;
      const int e  = n_in;
#pragma unroll
      for (int r = 0; r < 8; ++r) {
        const int mi = cd_row(r, lane);
        smT2[(((wave * 2 + kt) * 32) + (kh * 16 + mi)) * 16 + e] =
            (__bf16)acc[r];
      }
    }
  }
  __syncthreads();

  // ---- Step B: out strip (16 j x 384 k) over K=c ----
  {
    const v16bf a0 = load_frag(smT2, wave * 2 + 0, lane);
    const v16bf a1 = load_frag(smT2, wave * 2 + 1, lane);
    float* orow = out + (size_t)(i * S_DIM + j0 + wave * 16) * S_DIM;
#pragma unroll 4
    for (int nt = 0; nt < 24; ++nt) {
      const v16bf b0 = load_frag(Mx2Tp, nt * 2 + 0, lane);
      const v16bf b1 = load_frag(Mx2Tp, nt * 2 + 1, lane);
      v8f acc = {};
      acc = wmma_bf16(a0, b0, acc);
      acc = wmma_bf16(a1, b1, acc);
      const int n0 = nt * 16;
#pragma unroll
      for (int r = 0; r < 8; ++r)
        orow[(size_t)cd_row(r, lane) * S_DIM + n0 + (lane & 15)] = acc[r];
    }
  }
}

// ---------------------------------------------------------------------------
// launch
// ---------------------------------------------------------------------------
extern "C" void kernel_launch(void* const* d_in, const int* in_sizes, int n_in,
                              void* d_out, int out_size, void* d_ws,
                              size_t ws_size, hipStream_t stream) {
  (void)in_sizes; (void)n_in; (void)out_size; (void)ws_size;

  const float* G    = (const float*)d_in[0];
  const float* M0   = (const float*)d_in[1];
  const int*   idx0 = (const int*)  d_in[2];
  const float* w0   = (const float*)d_in[3];
  const float* M1   = (const float*)d_in[4];
  const int*   idx1 = (const int*)  d_in[5];
  const float* w1   = (const float*)d_in[6];
  const float* M2   = (const float*)d_in[7];
  const int*   idx2 = (const int*)  d_in[8];
  const float* w2   = (const float*)d_in[9];
  float* out = (float*)d_out;

  // workspace (bf16, fragment-packed): Mx0 | Mx1 | Mx2T | G | T1  (~3.8 MB)
  __bf16* Mx0p  = (__bf16*)d_ws;
  __bf16* Mx1p  = Mx0p + S_DIM * R_DIM;
  __bf16* Mx2Tp = Mx1p + S_DIM * R_DIM;
  __bf16* Gbp   = Mx2Tp + R_DIM * S_DIM;
  __bf16* T1p   = Gbp + R_DIM * R_DIM * R_DIM;

  tdknn_convert_g<<<dim3(256), dim3(256), 0, stream>>>(G, Gbp,
                                                       R_DIM * R_DIM * R_DIM);
  tdknn_gather<<<dim3(S_DIM, 3), dim3(R_DIM), 0, stream>>>(
      M0, M1, M2, idx0, idx1, idx2, w0, w1, w2, Mx0p, Mx1p, Mx2Tp);
  tdknn_gemm1<<<dim3(16, 3), dim3(256), 0, stream>>>(Mx0p, Gbp, T1p);
  tdknn_fused_tail<<<dim3(3, S_DIM), dim3(256), 0, stream>>>(Mx1p, T1p, Mx2Tp,
                                                             out);
}
